// NTMController_13554916786492
// MI455X (gfx1250) — matrix operand order repaired and checked by reading, not compiled, not run on hardware
//
#include <hip/hip_runtime.h>

// ---------------------------------------------------------------------------
// Types for CDNA5 WMMA (wave32)
// ---------------------------------------------------------------------------
typedef __attribute__((ext_vector_type(16))) __bf16 v16bf;
typedef __attribute__((ext_vector_type(8)))  __bf16 v8bf;
typedef __attribute__((ext_vector_type(8)))  float  v8f;

typedef unsigned short bf16s;   // bf16 storage (bit pattern)

__device__ __forceinline__ bf16s f32_to_bf16(float f) {
    unsigned int u = __float_as_uint(f);
    // round-to-nearest-even
    unsigned int r = (u + 0x7FFFu + ((u >> 16) & 1u)) >> 16;
    return (bf16s)r;
}

// ---------------------------------------------------------------------------
// Fragment loaders (CDNA5 ISA 7.12.2 layouts, wave32)
//
// A (16x32 bf16, M x K): lane l holds row m = l&15.
//   halves e=0..7  -> k = kBase + (l>>4)*8 + e
//   halves e=8..15 -> k = kBase + (l>>4)*8 + 16 + (e-8)
// B (32x16 bf16, K x N), stored as row-major [N][K] panel (== torch [out,in]):
//   lane l holds col n = l&15; same k mapping as A:
//   halves e=0..7 -> k = kBase + (l>>4)*8 + e, e=8..15 -> k = kBase + (l>>4)*8 + 16 + (e-8)
// ---------------------------------------------------------------------------
__device__ __forceinline__ v16bf load_a_frag(const bf16s* A, int lda,
                                             int mBase, int kBase, int lane) {
    const int m  = mBase + (lane & 15);
    const int kb = kBase + ((lane >> 4) << 3);
    const bf16s* p = A + (size_t)m * lda + kb;
    v8bf lo = *(const v8bf*)p;          // k = kb .. kb+7
    v8bf hi = *(const v8bf*)(p + 16);   // k = kb+16 .. kb+23
    return __builtin_shufflevector(lo, hi, 0,1,2,3,4,5,6,7,8,9,10,11,12,13,14,15);
}

__device__ __forceinline__ v16bf load_b_frag(const bf16s* Bp, int ldb,
                                             int nBase, int kBase, int lane) {
    const int n  = nBase + (lane & 15);
    const int kb = kBase + ((lane >> 4) << 3);
    const bf16s* p = Bp + (size_t)n * ldb + kb;
    v8bf lo = *(const v8bf*)p;
    v8bf hi = *(const v8bf*)(p + 16);
    return __builtin_shufflevector(lo, hi, 0,1,2,3,4,5,6,7,8,9,10,11,12,13,14,15);
}

// ---------------------------------------------------------------------------
// Generic bf16 WMMA GEMM:  C[64 x N] = A[64 x K] * B^T  (+ bias)
// B panel stored row-major [N][K].  One block computes 64 rows x 64 cols.
// 8 waves: wave = nPair(0..1) * 4 + mTile(0..3); each wave does two 16x16
// C tiles sharing its A fragment.   K % 32 == 0.
// ---------------------------------------------------------------------------
__global__ __launch_bounds__(256) void gemm_wmma_kernel(
    const bf16s* __restrict__ A, int lda,
    const bf16s* __restrict__ B, int ldb,
    float* __restrict__ C, int ldc,
    const float* __restrict__ bias, int K)
{
    const int lane  = threadIdx.x & 31;
    const int wave  = threadIdx.x >> 5;
    const int mTile = (wave & 3) * 16;
    const int nBase = blockIdx.x * 64 + (wave >> 2) * 32;

    v8f acc0 = {};
    v8f acc1 = {};

    for (int kk = 0; kk < K; kk += 32) {
        v16bf a  = load_a_frag(A, lda, mTile, kk, lane);
        v16bf b0 = load_b_frag(B, ldb, nBase,      kk, lane);
        v16bf b1 = load_b_frag(B, ldb, nBase + 16, kk, lane);
        acc0 = __builtin_amdgcn_wmma_f32_16x16x32_bf16(
                   false, a, false, b0, (short)0, acc0, false, false);
        acc1 = __builtin_amdgcn_wmma_f32_16x16x32_bf16(
                   false, a, false, b1, (short)0, acc1, false, false);
    }

    // C/D layout: VGPR r -> m = mTile + r + (lane>>4)*8 ; n = lane&15
    const int n  = lane & 15;
    const int mo = mTile + ((lane >> 4) << 3);
    const float bia0 = bias ? bias[nBase + n]      : 0.0f;
    const float bia1 = bias ? bias[nBase + 16 + n] : 0.0f;
#pragma unroll
    for (int r = 0; r < 8; ++r) {
        const int m = mo + r;
        C[(size_t)m * ldc + nBase + n]      = acc0[r] + bia0;
        C[(size_t)m * ldc + nBase + 16 + n] = acc1[r] + bia1;
    }
}

// ---------------------------------------------------------------------------
// Fused LSTM pointwise: gates -> (c,h); writes bf16 h into the gates A-panel
// (cols 64..319) and the dec A-panel (cols 0..255); packs x_{t_next} into
// gates A-panel cols 0..63.
// Grid: 80 blocks x 256 threads (64 LSTM blocks + 16 x-pack blocks).
// ---------------------------------------------------------------------------
__global__ __launch_bounds__(256) void lstm_pointwise_kernel(
    const float* __restrict__ gates,   // [64,1024]
    float* __restrict__ c,             // [64,256]
    bf16s* __restrict__ Abuf,          // [64,320] bf16
    bf16s* __restrict__ DecA,          // [64,384] bf16
    const float* __restrict__ x,       // [64,512,64]
    int t_next)
{
    const int blk = blockIdx.x;
    if (blk < 64) {
        const int e = blk * 256 + threadIdx.x;     // 0..16383
        const int b = e >> 8;
        const int j = e & 255;
        const float* g = gates + (size_t)b * 1024;
        const float gi = g[j];
        const float gf = g[256 + j];
        const float gg = g[512 + j];
        const float go = g[768 + j];
        const float si = 1.0f / (1.0f + __expf(-gi));
        const float sf = 1.0f / (1.0f + __expf(-gf));
        const float so = 1.0f / (1.0f + __expf(-go));
        const float cn = sf * c[e] + si * tanhf(gg);
        c[e] = cn;
        const float h = so * tanhf(cn);
        const bf16s hb = f32_to_bf16(h);
        Abuf[(size_t)b * 320 + 64 + j] = hb;
        DecA[(size_t)b * 384 + j]      = hb;
    } else {
        const int e = (blk - 64) * 256 + threadIdx.x;  // 0..4095
        const int b = e >> 6;
        const int k = e & 63;
        Abuf[(size_t)b * 320 + k] =
            f32_to_bf16(x[((size_t)b * 512 + t_next) * 64 + k]);
    }
}

// ---------------------------------------------------------------------------
// NTM memory step: one block per batch element, 128 threads.
//  phase 1: per-slot dot products with read/write keys
//  phase 2: two stable softmaxes via LDS tree reductions
//  phase 3: single coalesced pass over mem[b] computing read_vec and the
//           convex update  m = m*(1-ww[s]) + ww[s]*add[d]
// ---------------------------------------------------------------------------
__global__ __launch_bounds__(128) void memory_kernel(
    const float* __restrict__ keys,     // [64,256]  (read||write), no bias yet
    const float* __restrict__ b_read,   // [128]
    const float* __restrict__ b_write,  // [128]
    float* __restrict__ mem,            // [64,128,128]
    bf16s* __restrict__ DecA)           // [64,384] bf16 (cols 256.. = read_vec)
{
    __shared__ float rk[128];
    __shared__ float wk[128];
    __shared__ float s_rw[128];
    __shared__ float s_ww[128];
    __shared__ float red[128];

    const int b   = blockIdx.x;
    const int tid = threadIdx.x;

    rk[tid] = keys[(size_t)b * 256 + tid]       + b_read[tid];
    wk[tid] = keys[(size_t)b * 256 + 128 + tid] + b_write[tid];
    __syncthreads();

    // ---- phase 1: dots for slot s = tid -------------------------------
    const float* mrow = mem + ((size_t)b * 128 + tid) * 128;
    float dr = 0.0f, dw = 0.0f;
#pragma unroll 8
    for (int d = 0; d < 128; d += 4) {
        const float4 mv = *(const float4*)(mrow + d);
        dr += mv.x * rk[d] + mv.y * rk[d + 1] + mv.z * rk[d + 2] + mv.w * rk[d + 3];
        dw += mv.x * wk[d] + mv.y * wk[d + 1] + mv.z * wk[d + 2] + mv.w * wk[d + 3];
    }

    // ---- phase 2: softmaxes -------------------------------------------
    // max over dr
    red[tid] = dr; __syncthreads();
    for (int off = 64; off > 0; off >>= 1) {
        if (tid < off) red[tid] = fmaxf(red[tid], red[tid + off]);
        __syncthreads();
    }
    const float mr = red[0]; __syncthreads();
    const float er = __expf(dr - mr);
    red[tid] = er; __syncthreads();
    for (int off = 64; off > 0; off >>= 1) {
        if (tid < off) red[tid] += red[tid + off];
        __syncthreads();
    }
    const float sr = red[0]; __syncthreads();

    // max over dw
    red[tid] = dw; __syncthreads();
    for (int off = 64; off > 0; off >>= 1) {
        if (tid < off) red[tid] = fmaxf(red[tid], red[tid + off]);
        __syncthreads();
    }
    const float mw = red[0]; __syncthreads();
    const float ew = __expf(dw - mw);
    red[tid] = ew; __syncthreads();
    for (int off = 64; off > 0; off >>= 1) {
        if (tid < off) red[tid] += red[tid + off];
        __syncthreads();
    }
    const float sw = red[0]; __syncthreads();

    s_rw[tid] = er / sr;
    s_ww[tid] = ew / sw;
    rk[tid]   = tanhf(dw);      // add vector (S == D), reuse rk
    __syncthreads();

    // ---- phase 3: coalesced read_vec + update, thread owns column d ----
    const int d = tid;
    const float addd = rk[d];
    float rv = 0.0f;
    float* mcol = mem + (size_t)b * 128 * 128 + d;
#pragma unroll 4
    for (int s = 0; s < 128; ++s) {
        const float mo = mcol[(size_t)s * 128];
        rv += s_rw[s] * mo;
        mcol[(size_t)s * 128] = mo * (1.0f - s_ww[s]) + s_ww[s] * addd;
    }
    DecA[(size_t)b * 384 + 256 + d] = f32_to_bf16(rv);
}

// ---------------------------------------------------------------------------
// One-time prep: pack weights to bf16 [N][K] panels, fuse biases, zero c/mem,
// init A-panel with x_0 and h_0 = 0.
// ---------------------------------------------------------------------------
__global__ __launch_bounds__(256) void prep_kernel(
    const float* __restrict__ W_ih,    // [1024,64]
    const float* __restrict__ W_hh,    // [1024,256]
    const float* __restrict__ b_ih,    // [1024]
    const float* __restrict__ b_hh,    // [1024]
    const float* __restrict__ W_read,  // [128,256]
    const float* __restrict__ W_write, // [128,256]
    const float* __restrict__ W_dec,   // [64,384]
    const float* __restrict__ x,       // [64,512,64]
    bf16s* __restrict__ Wg,            // [1024,320]
    float* __restrict__ bg,            // [1024]
    bf16s* __restrict__ Wrw,           // [256,256]
    bf16s* __restrict__ Wdec,          // [64,384]
    bf16s* __restrict__ Abuf,          // [64,320]
    float* __restrict__ c,             // [64,256]
    float* __restrict__ mem)           // [64,128,128]
{
    long i = (long)blockIdx.x * 256 + threadIdx.x;

    if (i < 327680) {                 // Wg = [W_ih | W_hh] as [N][K] bf16
        const int n = (int)(i / 320), k = (int)(i % 320);
        const float v = (k < 64) ? W_ih[(size_t)n * 64 + k]
                                 : W_hh[(size_t)n * 256 + (k - 64)];
        Wg[i] = f32_to_bf16(v);
        return;
    }
    i -= 327680;
    if (i < 1024) { bg[i] = b_ih[i] + b_hh[i]; return; }
    i -= 1024;
    if (i < 65536) {                  // Wrw = [W_read ; W_write]
        const int n = (int)(i >> 8), k = (int)(i & 255);
        const float v = (n < 128) ? W_read[(size_t)n * 256 + k]
                                  : W_write[(size_t)(n - 128) * 256 + k];
        Wrw[i] = f32_to_bf16(v);
        return;
    }
    i -= 65536;
    if (i < 24576) { Wdec[i] = f32_to_bf16(W_dec[i]); return; }
    i -= 24576;
    if (i < 20480) {                  // Abuf = [x_0 | h_0=0]
        const int b = (int)(i / 320), j = (int)(i % 320);
        const float v = (j < 64) ? x[((size_t)b * 512) * 64 + j] : 0.0f;
        Abuf[i] = f32_to_bf16(v);
        return;
    }
    i -= 20480;
    if (i < 16384) { c[i] = 0.0f; return; }
    i -= 16384;
    if (i < 1048576) { mem[i] = 0.0f; return; }
}

// ---------------------------------------------------------------------------
// Host-side launch: one prep kernel + 512 * 5 kernels on `stream`.
// ---------------------------------------------------------------------------
extern "C" void kernel_launch(void* const* d_in, const int* in_sizes, int n_in,
                              void* d_out, int out_size, void* d_ws, size_t ws_size,
                              hipStream_t stream) {
    const float* x       = (const float*)d_in[0];
    const float* W_ih    = (const float*)d_in[1];
    const float* W_hh    = (const float*)d_in[2];
    const float* b_ih    = (const float*)d_in[3];
    const float* b_hh    = (const float*)d_in[4];
    const float* W_read  = (const float*)d_in[5];
    const float* b_read  = (const float*)d_in[6];
    const float* W_write = (const float*)d_in[7];
    const float* b_write = (const float*)d_in[8];
    const float* W_dec   = (const float*)d_in[9];
    const float* b_dec   = (const float*)d_in[10];

    char* ws = (char*)d_ws;
    // workspace layout (all offsets 4 KB aligned, >= 32B alignment for b128)
    bf16s* Wg    = (bf16s*)(ws + 0);         //  655360 B : [1024,320] bf16
    float* bg    = (float*)(ws + 655360);    //    4096 B : [1024] f32
    bf16s* Wrw   = (bf16s*)(ws + 659456);    //  131072 B : [256,256] bf16
    bf16s* Wdec  = (bf16s*)(ws + 790528);    //   49152 B : [64,384] bf16
    bf16s* Abuf  = (bf16s*)(ws + 839680);    //   40960 B : [64,320] bf16
    bf16s* DecA  = (bf16s*)(ws + 880640);    //   49152 B : [64,384] bf16
    float* gates = (float*)(ws + 929792);    //  262144 B : [64,1024] f32
    float* keys  = (float*)(ws + 1191936);   //   65536 B : [64,256] f32
    float* c     = (float*)(ws + 1257472);   //   65536 B : [64,256] f32
    float* mem   = (float*)(ws + 1323008);   // 4194304 B : [64,128,128] f32

    float* out = (float*)d_out;              // [64,512,64]

    prep_kernel<<<5876, 256, 0, stream>>>(W_ih, W_hh, b_ih, b_hh, W_read,
                                          W_write, W_dec, x,
                                          Wg, bg, Wrw, Wdec, Abuf, c, mem);

    for (int t = 0; t < 512; ++t) {
        // gates = [x_t|h] @ [W_ih|W_hh]^T + (b_ih+b_hh)   (M=64,N=1024,K=320)
        gemm_wmma_kernel<<<16, 256, 0, stream>>>(Abuf, 320, Wg, 320,
                                                 gates, 1024, bg, 320);
        // LSTM pointwise, write bf16 h panels, pack x_{t+1}
        lstm_pointwise_kernel<<<80, 256, 0, stream>>>(gates, c, Abuf, DecA, x,
                                                      (t < 511) ? t + 1 : 511);
        // keys = h @ [W_read;W_write]^T                    (M=64,N=256,K=256)
        gemm_wmma_kernel<<<4, 256, 0, stream>>>(Abuf + 64, 320, Wrw, 256,
                                                keys, 256, nullptr, 256);
        // NTM addressing + memory update + read_vec
        memory_kernel<<<64, 128, 0, stream>>>(keys, b_read, b_write, mem, DecA);
        // out[:,t,:] = [h|read_vec] @ W_dec^T + b_dec      (M=64,N=64,K=384)
        gemm_wmma_kernel<<<1, 256, 0, stream>>>(DecA, 384, Wdec, 384,
                                                out + (size_t)t * 64, 512 * 64,
                                                b_dec, 384);
    }
}
